// SparseGCNLayer_40321152975477
// MI455X (gfx1250) — matrix operand order repaired and checked by reading, not compiled, not run on hardware
//
#include <hip/hip_runtime.h>
#include <stdint.h>

#define Bb 16
#define Nn 1000
#define SFc 20
#define Hh 128
#define Ee 20000
#define MNODE (Bb*Nn)     // 16000 rows
#define MEDGE (Bb*Ee)     // 320000 rows
#define PITCH 136         // LDS row pitch in bf16 elems: 16B-aligned chunks, 4-bank lane stride
#define EPSV 1e-5f

typedef __attribute__((ext_vector_type(16))) __bf16          bf16x16;
typedef __attribute__((ext_vector_type(8)))  float           f32x8;
typedef __attribute__((ext_vector_type(8)))  unsigned short  u16x8;

union AFrag { bf16x16 v; u16x8 h[2]; __bf16 b[16]; };

// native converts: clang legalizes these to gfx1250 bf16 cvt instructions
__device__ __forceinline__ __bf16 f2bf(float f) { return (__bf16)f; }
__device__ __forceinline__ float  bf2f(__bf16 h) { return (float)h; }

__device__ __forceinline__ f32x8 wmma_bf16(bf16x16 a, bf16x16 b, f32x8 c) {
  return __builtin_amdgcn_wmma_f32_16x16x32_bf16(false, a, false, b, (short)0, c, false, false);
}

// ---- B fragment from LDS: Wt is n-major bf16, Wt[n*PITCH + k] = W[k][n] ----
// 16-bit B 32x16 layout: lane<16 -> col = lane, K = {0..7, 16..23}+kt*32;
//                        lane>=16 -> col = lane-16, K = {8..15, 24..31}+kt*32.
__device__ __forceinline__ bf16x16 frag_b_lds(const __bf16* Wt, int nt, int kt, int lane) {
  int c  = nt*16 + (lane & 15);
  int k0 = kt*32 + ((lane < 16) ? 0 : 8);
  AFrag f;
  f.h[0] = *(const u16x8*)(Wt + c*PITCH + k0);
  f.h[1] = *(const u16x8*)(Wt + c*PITCH + k0 + 16);
  return f.v;
}

// ---- A fragment streamed from global f32 rows with on-the-fly bf16 convert ----
// 16-bit A 16x32 layout: lane<16 -> row=lane, K = {0..7, 16..23}+kt*32;
//                        lane>=16 -> row=lane-16, K = {8..15, 24..31}+kt*32.
__device__ __forceinline__ bf16x16 frag_a_glb(const float* rowBase, int kt, int lane) {
  int r  = lane & 15;
  int k0 = kt*32 + ((lane < 16) ? 0 : 8);
  const float* p = rowBase + (size_t)r*Hh + k0;
  AFrag f;
#pragma unroll
  for (int i = 0; i < 8; ++i) f.b[i]   = f2bf(p[i]);
#pragma unroll
  for (int i = 0; i < 8; ++i) f.b[8+i] = f2bf(p[16+i]);
  return f.v;
}

__device__ __forceinline__ bf16x16 frag_a_glb_clamp(const float* base0, int mt, int rows,
                                                    int kt, int lane) {
  int r = mt*16 + (lane & 15);
  AFrag f;
  if (r < rows) {
    int k0 = kt*32 + ((lane < 16) ? 0 : 8);
    const float* p = base0 + (size_t)r*Hh + k0;
#pragma unroll
    for (int i = 0; i < 8; ++i) f.b[i]   = f2bf(p[i]);
#pragma unroll
    for (int i = 0; i < 8; ++i) f.b[8+i] = f2bf(p[16+i]);
  } else {
#pragma unroll
    for (int i = 0; i < 16; ++i) f.b[i] = (__bf16)0.0f;
  }
  return f.v;
}

// stage f32 row-major weight W[k][n] -> LDS Wt[n*PITCH+k] (bf16)
__device__ __forceinline__ void stage_w(const float* W, __bf16* Wt, int tid, int nthr) {
  for (int i = tid; i < Hh*Hh; i += nthr) {
    int k = i >> 7, n = i & 127;
    Wt[n*PITCH + k] = f2bf(W[i]);
  }
}

// ======================= generic M x 128 x 128 GEMM =======================
// mode 0: f32 out ; mode 1: bf16 out ; mode 2: bf16 out remapped to per-batch (E+1)-row table
__global__ __launch_bounds__(256)
void gemm128(const float* __restrict__ X, const float* __restrict__ W,
             const float* __restrict__ bias, void* __restrict__ out, int mode) {
  __shared__ __bf16 Wt[Hh*PITCH];
  int tid = threadIdx.x;
  stage_w(W, Wt, tid, 256);
  __syncthreads();
  int wave = tid >> 5, lane = tid & 31;
  size_t g0 = (size_t)blockIdx.x*128 + (size_t)wave*16;
  const float* rowBase = X + g0*Hh;
  f32x8 acc[8] = {};
#pragma unroll
  for (int kt = 0; kt < 4; ++kt) {
    bf16x16 a = frag_a_glb(rowBase, kt, lane);
#pragma unroll
    for (int nt = 0; nt < 8; ++nt) {
      bf16x16 b = frag_b_lds(Wt, nt, kt, lane);
      acc[nt] = wmma_bf16(a, b, acc[nt]);
    }
  }
  int half = (lane < 16) ? 0 : 8;
#pragma unroll
  for (int nt = 0; nt < 8; ++nt) {
    int col = nt*16 + (lane & 15);
    float bv = bias[col];
#pragma unroll
    for (int r = 0; r < 8; ++r) {
      size_t row = g0 + (size_t)(r + half);
      float v = acc[nt][r] + bv;
      if (mode == 0)      ((float*)out)[row*Hh + col] = v;
      else if (mode == 1) ((__bf16*)out)[row*Hh + col] = f2bf(v);
      else {
        size_t orow = row + row/(size_t)Ee;   // insert placeholder row per batch
        ((__bf16*)out)[orow*Hh + col] = f2bf(v);
      }
    }
  }
}

// write W_ph (bf16) to row E of each batch of the iUe table
__global__ void write_ph(const float* __restrict__ Wph, __bf16* __restrict__ iUeBf) {
  size_t b = blockIdx.x;
  iUeBf[(b*(size_t)(Ee+1) + Ee)*Hh + threadIdx.x] = f2bf(Wph[threadIdx.x]);
}

// ======================= node aggregation (softmax attention) =======================
// 4 waves/block, one node each: Ve = softmax_s(e_tile @ ee_w + b); x_tmp = Ux + sum_s Ve*Vx[gather]
__global__ __launch_bounds__(128)
void node_fuse(const float* __restrict__ eT, const int* __restrict__ eidxG,
               const float* __restrict__ eeW, const float* __restrict__ eeB,
               const __bf16* __restrict__ VxBf, const float* __restrict__ Ux,
               float* __restrict__ xtmp) {
  __shared__ __bf16 Wt[Hh*PITCH];
  __shared__ __bf16 Vxg[4][SFc*PITCH];
  __shared__ int eidx[4][SFc];
  int tid = threadIdx.x, wave = tid >> 5, lane = tid & 31;
  stage_w(eeW, Wt, tid, 128);
  int g = blockIdx.x*4 + wave;
  int b = g / Nn, n = g - b*Nn;
  size_t eBase = (size_t)b*Ee + (size_t)n*SFc;
  const float* eRows = eT + eBase*Hh;
  if (lane < SFc) eidx[wave][lane] = eidxG[eBase + lane];
  __syncthreads();
  for (int i = lane; i < SFc*Hh; i += 32) {            // stage gathered Vx rows (bf16)
    int s = i >> 7, k = i & 127;
    size_t srow = (size_t)b*Nn + (size_t)eidx[wave][s];
    Vxg[wave][s*PITCH + k] = VxBf[srow*Hh + k];
  }
  __syncthreads();

  f32x8 acc0[8] = {}; f32x8 acc1[8] = {};
#pragma unroll
  for (int kt = 0; kt < 4; ++kt) {
    bf16x16 a0 = frag_a_glb_clamp(eRows, 0, SFc, kt, lane);
    bf16x16 a1 = frag_a_glb_clamp(eRows, 1, SFc, kt, lane);
#pragma unroll
    for (int nt = 0; nt < 8; ++nt) {
      bf16x16 bf = frag_b_lds(Wt, nt, kt, lane);
      acc0[nt] = wmma_bf16(a0, bf, acc0[nt]);
      acc1[nt] = wmma_bf16(a1, bf, acc1[nt]);
    }
  }
  bool lo = lane < 16;
#pragma unroll
  for (int nt = 0; nt < 8; ++nt) {
    int col = nt*16 + (lane & 15);
    float bb = eeB[col];
    float v0[8], v1[4];
    float mloc = -3.0e38f;
#pragma unroll
    for (int r = 0; r < 8; ++r) { v0[r] = acc0[nt][r] + bb; mloc = fmaxf(mloc, v0[r]); }
    if (lo) {
#pragma unroll
      for (int r = 0; r < 4; ++r) { v1[r] = acc1[nt][r] + bb; mloc = fmaxf(mloc, v1[r]); }
    }
    float m = fmaxf(mloc, __shfl_xor(mloc, 16, 32));   // pair lanes l, l^16 (same column)
    float ssum = 0.f;
#pragma unroll
    for (int r = 0; r < 8; ++r) { v0[r] = __expf(v0[r] - m); ssum += v0[r]; }
    if (lo) {
#pragma unroll
      for (int r = 0; r < 4; ++r) { v1[r] = __expf(v1[r] - m); ssum += v1[r]; }
    }
    ssum += __shfl_xor(ssum, 16, 32);
    float inv = 1.0f / ssum;
    float agg = 0.f;
#pragma unroll
    for (int r = 0; r < 8; ++r) {
      int s = lo ? r : 8 + r;
      agg += v0[r] * bf2f(Vxg[wave][s*PITCH + col]);
    }
    if (lo) {
#pragma unroll
      for (int r = 0; r < 4; ++r) agg += v1[r] * bf2f(Vxg[wave][(16 + r)*PITCH + col]);
    }
    agg *= inv;
    agg += __shfl_xor(agg, 16, 32);
    if (lo) xtmp[(size_t)g*Hh + col] = Ux[(size_t)g*Hh + col] + agg;
  }
}

// ======================= edge feature assembly =======================
__global__ __launch_bounds__(128)
void edge_fuse(const float* __restrict__ eT, const int* __restrict__ eidxG,
               const int* __restrict__ invIdxG,
               const float* __restrict__ Uw, const float* __restrict__ Ub,
               const __bf16* __restrict__ VxtBf, const float* __restrict__ Vxf,
               const __bf16* __restrict__ iUeBf, float* __restrict__ etmp) {
  __shared__ __bf16 Wt[Hh*PITCH];
  __shared__ __bf16 Vtg[4][SFc*PITCH];
  __shared__ int eidx[4][SFc];
  __shared__ int iidx[4][SFc];
  int tid = threadIdx.x, wave = tid >> 5, lane = tid & 31;
  stage_w(Uw, Wt, tid, 128);
  int g = blockIdx.x*4 + wave;
  int b = g / Nn, n = g - b*Nn;
  size_t eBase = (size_t)b*Ee + (size_t)n*SFc;
  const float* eRows = eT + eBase*Hh;
  if (lane < SFc) { eidx[wave][lane] = eidxG[eBase + lane]; iidx[wave][lane] = invIdxG[eBase + lane]; }
  __syncthreads();
  for (int i = lane; i < SFc*Hh; i += 32) {
    int s = i >> 7, k = i & 127;
    size_t srow = (size_t)b*Nn + (size_t)eidx[wave][s];
    Vtg[wave][s*PITCH + k] = VxtBf[srow*Hh + k];
  }
  __syncthreads();

  f32x8 acc0[8] = {}; f32x8 acc1[8] = {};
#pragma unroll
  for (int kt = 0; kt < 4; ++kt) {
    bf16x16 a0 = frag_a_glb_clamp(eRows, 0, SFc, kt, lane);
    bf16x16 a1 = frag_a_glb_clamp(eRows, 1, SFc, kt, lane);
#pragma unroll
    for (int nt = 0; nt < 8; ++nt) {
      bf16x16 bf = frag_b_lds(Wt, nt, kt, lane);
      acc0[nt] = wmma_bf16(a0, bf, acc0[nt]);
      acc1[nt] = wmma_bf16(a1, bf, acc1[nt]);
    }
  }
  bool lo = lane < 16;
  size_t iBase = (size_t)b*(size_t)(Ee + 1);
#pragma unroll
  for (int nt = 0; nt < 8; ++nt) {
    int col = nt*16 + (lane & 15);
    float bv = Ub[col] + Vxf[(size_t)g*Hh + col];      // Ue bias + broadcast Vxf row
#pragma unroll
    for (int r = 0; r < 8; ++r) {
      int s = lo ? r : 8 + r;
      float val = acc0[nt][r] + bv
                + bf2f(Vtg[wave][s*PITCH + col])
                + bf2f(iUeBf[(iBase + (size_t)iidx[wave][s])*Hh + col]);
      etmp[(eBase + (size_t)s)*Hh + col] = val;
    }
    if (lo) {
#pragma unroll
      for (int r = 0; r < 4; ++r) {
        int s = 16 + r;
        float val = acc1[nt][r] + bv
                  + bf2f(Vtg[wave][s*PITCH + col])
                  + bf2f(iUeBf[(iBase + (size_t)iidx[wave][s])*Hh + col]);
        etmp[(eBase + (size_t)s)*Hh + col] = val;
      }
    }
  }
}

// ======================= deterministic BN stats + finalize =======================
__global__ __launch_bounds__(128)
void col_stats_partial(const float* __restrict__ buf, float* __restrict__ partial,
                       int rowsPerBlock, long long totalRows) {
  int c = threadIdx.x;
  long long r0 = (long long)blockIdx.x * rowsPerBlock;
  long long r1 = r0 + rowsPerBlock; if (r1 > totalRows) r1 = totalRows;
  float s = 0.f, q = 0.f;
  for (long long r = r0; r < r1; ++r) { float v = buf[(size_t)r*Hh + c]; s += v; q += v*v; }
  partial[(size_t)blockIdx.x*256 + c]       = s;
  partial[(size_t)blockIdx.x*256 + 128 + c] = q;
}
__global__ __launch_bounds__(256)
void col_stats_final(const float* __restrict__ partial, int nblk, float* __restrict__ out) {
  int t = threadIdx.x;
  float s = 0.f;
  for (int i = 0; i < nblk; ++i) s += partial[(size_t)i*256 + t];  // fixed order
  out[t] = s;
}
__global__ __launch_bounds__(256)
void bn_residual(const float* __restrict__ orig, float* __restrict__ io,
                 const float* __restrict__ gstats, const float* __restrict__ gamma,
                 const float* __restrict__ beta, float invCount) {
  size_t i = (size_t)blockIdx.x*256 + threadIdx.x;
  int col = (int)(i & 127);
  float mean = gstats[col] * invCount;
  float var  = gstats[128 + col] * invCount - mean*mean;
  float y = (io[i] - mean) * rsqrtf(var + EPSV) * gamma[col] + beta[col];
  io[i] = orig[i] + fmaxf(y, 0.f);
}

extern "C" void kernel_launch(void* const* d_in, const int* in_sizes, int n_in,
                              void* d_out, int out_size, void* d_ws, size_t ws_size,
                              hipStream_t stream) {
  const float* x    = (const float*)d_in[0];
  const float* e    = (const float*)d_in[1];
  const int* eidx   = (const int*)d_in[2];
  const int* invidx = (const int*)d_in[3];
  const float* node_w = (const float*)d_in[4];  const float* node_b = (const float*)d_in[5];
  const float* to_w   = (const float*)d_in[6];  const float* to_b   = (const float*)d_in[7];
  const float* ee_w   = (const float*)d_in[8];  const float* ee_b   = (const float*)d_in[9];
  const float* U_w    = (const float*)d_in[10]; const float* U_b    = (const float*)d_in[11];
  const float* Vf_w   = (const float*)d_in[12]; const float* Vf_b   = (const float*)d_in[13];
  const float* Vt_w   = (const float*)d_in[14]; const float* Vt_b   = (const float*)d_in[15];
  const float* iU_w   = (const float*)d_in[16]; const float* iU_b   = (const float*)d_in[17];
  const float* W_ph   = (const float*)d_in[18];
  const float* bn_ng  = (const float*)d_in[19]; const float* bn_nb  = (const float*)d_in[20];
  const float* bn_eg  = (const float*)d_in[21]; const float* bn_eb  = (const float*)d_in[22];

  char* ws = (char*)d_ws;
  size_t off = 0;
  float* statsN = (float*)(ws + off); off += 256*sizeof(float);
  float* statsE = (float*)(ws + off); off += 256*sizeof(float);
  off = (off + 255) & ~(size_t)255;
  float* partials = (float*)(ws + off); off += (size_t)256*256*sizeof(float);   // max 256 blocks
  float* Ux  = (float*)(ws + off); off += (size_t)MNODE*Hh*sizeof(float);
  float* Vxf = (float*)(ws + off); off += (size_t)MNODE*Hh*sizeof(float);
  __bf16* VxBf  = (__bf16*)(ws + off); off += (size_t)MNODE*Hh*2;
  __bf16* VxtBf = (__bf16*)(ws + off); off += (size_t)MNODE*Hh*2;
  __bf16* iUeBf = (__bf16*)(ws + off); off += (size_t)Bb*(Ee+1)*Hh*2;

  float* xout = (float*)d_out;                     // x_new region (also x_tmp)
  float* eout = xout + (size_t)MNODE*Hh;           // e_new region (also e_tmp)

  // ---- NodeFeatures ----
  gemm128<<<MNODE/128, 256, 0, stream>>>(x, node_w, node_b, (void*)Ux, 0);
  gemm128<<<MNODE/128, 256, 0, stream>>>(x, to_w,   to_b,   (void*)VxBf, 1);
  gemm128<<<MEDGE/128, 256, 0, stream>>>(e, iU_w,   iU_b,   (void*)iUeBf, 2);
  write_ph<<<Bb, Hh, 0, stream>>>(W_ph, iUeBf);
  node_fuse<<<MNODE/4, 128, 0, stream>>>(e, eidx, ee_w, ee_b, VxBf, Ux, xout);
  col_stats_partial<<<64, 128, 0, stream>>>(xout, partials, 250, (long long)MNODE);
  col_stats_final<<<1, 256, 0, stream>>>(partials, 64, statsN);
  bn_residual<<<(MNODE*Hh)/256, 256, 0, stream>>>(x, xout, statsN, bn_ng, bn_nb, 1.0f/MNODE);
  // ---- EdgeFeatures ----
  gemm128<<<MNODE/128, 256, 0, stream>>>(xout, Vf_w, Vf_b, (void*)Vxf, 0);
  gemm128<<<MNODE/128, 256, 0, stream>>>(xout, Vt_w, Vt_b, (void*)VxtBf, 1);
  edge_fuse<<<MNODE/4, 128, 0, stream>>>(e, eidx, invidx, U_w, U_b, VxtBf, Vxf, iUeBf, eout);
  col_stats_partial<<<256, 128, 0, stream>>>(eout, partials, 1250, (long long)MEDGE);
  col_stats_final<<<1, 256, 0, stream>>>(partials, 256, statsE);
  bn_residual<<<160000, 256, 0, stream>>>(e, eout, statsE, bn_eg, bn_eb, 1.0f/MEDGE);
}
